// SF_DPL_Model_87522843558222
// MI455X (gfx1250) — matrix operand order repaired and checked by reading, not compiled, not run on hardware
//
#include <hip/hip_runtime.h>
#include <hip/hip_bf16.h>

#define B_   512
#define R_   116
#define FIN_ 116
#define H_   64
#define EPG_ 3712
#define N_   (B_ * R_)     // 59392
#define E_   (B_ * EPG_)   // 1900544
#define EPSV 1e-5f

typedef __attribute__((ext_vector_type(16))) _Float16 v16h;
typedef __attribute__((ext_vector_type(8)))  _Float16 v8h;
typedef __attribute__((ext_vector_type(8)))  float    v8f;

union F16x16 { v16h v; v8h h[2]; };

__device__ __forceinline__ void atomAdd(float* p, float v) {
    (void)__hip_atomic_fetch_add(p, v, __ATOMIC_RELAXED, __HIP_MEMORY_SCOPE_AGENT);
}

// ---------------- degree / norm ----------------
__global__ void deg_init(float* __restrict__ deg) {
    deg[blockIdx.x * 256 + threadIdx.x] = 1.0f;   // self-loop weight
}
__global__ void deg_accum(const int* __restrict__ dst, const float* __restrict__ w,
                          float* __restrict__ deg) {
    int e = blockIdx.x * 256 + threadIdx.x;
    atomAdd(&deg[dst[e]], w[e]);
}
__global__ void deg_finish(float* __restrict__ deg) {
    int i = blockIdx.x * 256 + threadIdx.x;
    deg[i] = rsqrtf(deg[i]);                      // deg >= 1 always
}
__global__ void zero_stats(float* __restrict__ st1, float* __restrict__ st2) {
    int t = threadIdx.x;                          // 128 threads: sum+sumsq of both layers
    st1[t] = 0.f; st2[t] = 0.f;
}

// ---------------- WMMA GEMM: out[32 rows x 64 cols] per 256-thread block ----------------
// A: [N, KREAL] fp32 (+optional per-ROI prompt, +optional fused BN affine+ReLU)
// W: [KREAL, 64] fp32 ; out: [N, 64] fp32
template <int KREAL, int KPAD, bool PROMPT, bool AFFINE>
__global__ void __launch_bounds__(256)
gemm32_wmma(const float* __restrict__ A, const float* __restrict__ prompt,
            const float* __restrict__ W,
            const float* __restrict__ aff_a, const float* __restrict__ aff_b,
            float* __restrict__ out)
{
    __shared__ _Float16 As[32][136];   // padded stride (272B) -> conflict-free frag loads
    __shared__ _Float16 Wt[64][136];   // W transposed: Wt[col][k]

    const int tid = threadIdx.x;
    const int rowBase = blockIdx.x * 32;

    // stage A tile (fp32 -> f16, with prompt / fused BN affine+ReLU)
    for (int i = tid; i < 32 * KPAD; i += 256) {
        int r = i / KPAD, k = i % KPAD;
        float v = 0.f;
        if (k < KREAL) {
            int n = rowBase + r;
            v = A[n * KREAL + k];
            if (PROMPT) v += prompt[(n % R_) * KREAL + k];
            if (AFFINE) v = fmaxf(aff_a[k] * v + aff_b[k], 0.f);
        }
        As[r][k] = (_Float16)v;
    }
    // stage W transposed (consecutive tids write consecutive k of one column)
    for (int i = tid; i < 64 * KPAD; i += 256) {
        int c = i / KPAD, k = i % KPAD;
        Wt[c][k] = (_Float16)((k < KREAL) ? W[k * 64 + c] : 0.f);
    }
    __syncthreads();

    const int wave   = tid >> 5;
    const int lane   = tid & 31;
    const int rowOff = (wave >> 2) * 16;          // which 16-row half of the tile
    const int mrow   = rowOff + (lane & 15);
    const int khalf  = lane >> 4;
    const int colBase = (wave & 3) * 16;
    const int colN   = lane & 15;

    v8f c = {};
    #pragma unroll
    for (int kc = 0; kc < KPAD; kc += 32) {
        F16x16 a, b;
        // A 16x32 f16 layout: lane half picks K {0..7,16..23} / {8..15,24..31}
        a.h[0] = *(const v8h*)&As[mrow][kc + khalf * 8];
        a.h[1] = *(const v8h*)&As[mrow][kc + 16 + khalf * 8];
        // B 32x16 f16 layout: lane half picks K {0..7,16..23} / {8..15,24..31} (matches A)
        b.h[0] = *(const v8h*)&Wt[colBase + colN][kc + khalf * 8];
        b.h[1] = *(const v8h*)&Wt[colBase + colN][kc + 16 + khalf * 8];
        c = __builtin_amdgcn_wmma_f32_16x16x32_f16(false, a.v, false, b.v,
                                                   (short)0, c, false, false);
    }

    // C/D f32 layout: VGPR r -> M = r (+8 for lanes 16-31), N = lane&15
    const int mOff = khalf * 8;
    #pragma unroll
    for (int r = 0; r < 8; ++r)
        out[(rowBase + rowOff + mOff + r) * 64 + colBase + colN] = c[r];
}

// ---------------- GCN aggregation ----------------
// h[n,f] = dinv[n]^2 * t[n,f] + bias[f]   (self-loop term + bias)
__global__ void init_h(const float* __restrict__ t, const float* __restrict__ dinv,
                       const float* __restrict__ bias, float* __restrict__ h) {
    int idx = blockIdx.x * 256 + threadIdx.x;
    int n = idx >> 6, f = idx & 63;
    float d = dinv[n];
    h[idx] = d * d * t[idx] + bias[f];
}
// h[dst,f] += dinv[src]*w*dinv[dst] * t[src,f]
__global__ void edge_agg(const float* __restrict__ t, const int* __restrict__ src,
                         const int* __restrict__ dst, const float* __restrict__ w,
                         const float* __restrict__ dinv, float* __restrict__ h) {
    int idx = blockIdx.x * 256 + threadIdx.x;
    int e = idx >> 6, f = idx & 63;
    int s = src[e], d = dst[e];
    float norm = dinv[s] * w[e] * dinv[d];
    atomAdd(&h[d * 64 + f], norm * t[s * 64 + f]);
}

// ---------------- BatchNorm (fused affine) ----------------
__global__ void bn_stats(const float* __restrict__ h, float* __restrict__ st) {
    int f = threadIdx.x;                 // 64 threads
    int row0 = blockIdx.x * 256;
    float s = 0.f, ss = 0.f;
    for (int r = 0; r < 256; ++r) {
        float v = h[(row0 + r) * 64 + f];
        s += v; ss += v * v;
    }
    atomAdd(&st[f], s);
    atomAdd(&st[64 + f], ss);
}
__global__ void bn_finalize(float* __restrict__ st, const float* __restrict__ g,
                            const float* __restrict__ be) {
    int f = threadIdx.x;                 // 64 threads
    float mu   = st[f] * (1.0f / (float)N_);
    float var  = st[64 + f] * (1.0f / (float)N_) - mu * mu;
    float rstd = rsqrtf(var + EPSV);
    st[128 + f] = g[f] * rstd;                 // scale
    st[192 + f] = be[f] - g[f] * mu * rstd;    // shift
}

// ---------------- pooling (applies BN2 affine + ReLU) ----------------
__global__ void pool_kernel(const float* __restrict__ h, const float* __restrict__ st,
                            float* __restrict__ z) {
    int b = blockIdx.x, f = threadIdx.x;       // 512 x 64
    float a = st[128 + f], bb = st[192 + f];
    float s = 0.f, mx = -1e30f;
    for (int r = 0; r < R_; ++r) {
        float v = a * h[(b * R_ + r) * 64 + f] + bb;
        v = v > 0.f ? v : 0.f;
        s += v; mx = fmaxf(mx, v);
    }
    z[b * 128 + f]      = s * (1.0f / (float)R_);
    z[b * 128 + 64 + f] = mx;
}

// ---------------- MLP head ----------------
__global__ void mlp_kernel(const float* __restrict__ z,
                           const float* __restrict__ Wc1, const float* __restrict__ bc1,
                           const float* __restrict__ Wc2, const float* __restrict__ bc2,
                           const float* __restrict__ Wc3, const float* __restrict__ bc3,
                           float* __restrict__ out) {
    __shared__ float zs[128], m1[64], m2[32];
    int b = blockIdx.x, t = threadIdx.x;       // 512 x 64
    zs[t]      = z[b * 128 + t];
    zs[t + 64] = z[b * 128 + 64 + t];
    __syncthreads();
    float acc = bc1[t];
    for (int i = 0; i < 128; ++i) acc += zs[i] * Wc1[i * 64 + t];
    m1[t] = acc > 0.f ? acc : 0.f;
    __syncthreads();
    if (t < 32) {
        float a2 = bc2[t];
        for (int i = 0; i < 64; ++i) a2 += m1[i] * Wc2[i * 32 + t];
        m2[t] = a2 > 0.f ? a2 : 0.f;
    }
    __syncthreads();
    if (t < 2) {
        float a3 = bc3[t];
        for (int i = 0; i < 32; ++i) a3 += m2[i] * Wc3[i * 2 + t];
        out[b * 2 + t] = a3;
    }
}

extern "C" void kernel_launch(void* const* d_in, const int* in_sizes, int n_in,
                              void* d_out, int out_size, void* d_ws, size_t ws_size,
                              hipStream_t stream) {
    (void)in_sizes; (void)n_in; (void)out_size; (void)ws_size;
    const float* x      = (const float*)d_in[0];
    const int*   ei     = (const int*)  d_in[1];
    const float* ew     = (const float*)d_in[2];
    // d_in[3] = batch (implicit from layout; unused)
    const float* prompt = (const float*)d_in[4];
    const float* W1  = (const float*)d_in[5];
    const float* b1  = (const float*)d_in[6];
    const float* g1  = (const float*)d_in[7];
    const float* be1 = (const float*)d_in[8];
    const float* W2  = (const float*)d_in[9];
    const float* b2  = (const float*)d_in[10];
    const float* g2  = (const float*)d_in[11];
    const float* be2 = (const float*)d_in[12];
    const float* Wc1 = (const float*)d_in[13];
    const float* bc1 = (const float*)d_in[14];
    const float* Wc2 = (const float*)d_in[15];
    const float* bc2 = (const float*)d_in[16];
    const float* Wc3 = (const float*)d_in[17];
    const float* bc3 = (const float*)d_in[18];
    float* out = (float*)d_out;

    const int* src = ei;
    const int* dst = ei + E_;

    float* ws   = (float*)d_ws;
    float* dinv = ws;                    // N_
    float* t    = dinv + N_;             // N_*64
    float* h    = t + (size_t)N_ * 64;   // N_*64
    float* st1  = h + (size_t)N_ * 64;   // 256 (sum,sumsq,a,b)
    float* st2  = st1 + 256;             // 256
    float* z    = st2 + 256;             // B_*128

    // degrees + symmetric norm
    deg_init  <<<N_ / 256, 256, 0, stream>>>(dinv);
    deg_accum <<<E_ / 256, 256, 0, stream>>>(dst, ew, dinv);
    deg_finish<<<N_ / 256, 256, 0, stream>>>(dinv);
    zero_stats<<<1, 128, 0, stream>>>(st1, st2);

    // layer 1: (x + prompt) @ W1  -> aggregate -> BN stats
    gemm32_wmma<FIN_, 128, true, false><<<N_ / 32, 256, 0, stream>>>(
        x, prompt, W1, nullptr, nullptr, t);
    init_h  <<<(N_ * 64) / 256, 256, 0, stream>>>(t, dinv, b1, h);
    edge_agg<<<(E_ * 64) / 256, 256, 0, stream>>>(t, src, dst, ew, dinv, h);
    bn_stats<<<N_ / 256, 64, 0, stream>>>(h, st1);
    bn_finalize<<<1, 64, 0, stream>>>(st1, g1, be1);

    // layer 2: relu(bn1(h)) @ W2 (affine+relu fused into A-load) -> aggregate -> BN stats
    gemm32_wmma<H_, 64, false, true><<<N_ / 32, 256, 0, stream>>>(
        h, nullptr, W2, st1 + 128, st1 + 192, t);
    init_h  <<<(N_ * 64) / 256, 256, 0, stream>>>(t, dinv, b2, h);
    edge_agg<<<(E_ * 64) / 256, 256, 0, stream>>>(t, src, dst, ew, dinv, h);
    bn_stats<<<N_ / 256, 64, 0, stream>>>(h, st2);
    bn_finalize<<<1, 64, 0, stream>>>(st2, g2, be2);

    // pool (bn2 affine + relu fused) + MLP head
    pool_kernel<<<B_, 64, 0, stream>>>(h, st2, z);
    mlp_kernel <<<B_, 64, 0, stream>>>(z, Wc1, bc1, Wc2, bc2, Wc3, bc3, out);
}
